// Block_29609504539559
// MI455X (gfx1250) — matrix operand order repaired and checked by reading, not compiled, not run on hardware
//
#include <hip/hip_runtime.h>
#include <hip/hip_bf16.h>

// ---------------- problem constants ----------------
#define BATCH   4
#define CIN     13          // input channels
#define CC      31          // conv input channels (13 + 18 tensor-product)
#define KPAD    32          // CC padded to 32 for WMMA K
#define DD      48
#define DP      54          // 48 + 2*3 halo
#define NB      75          // basis count
#define NTAP    343         // 7^3
#define PLANE   (DD*DD*DD)  // 110592 voxels per channel-plane
#define EPS     1e-5f

typedef __attribute__((ext_vector_type(16))) _Float16 v16h;
typedef __attribute__((ext_vector_type(8)))  float    v8f;

#define WMMA_F16(a, b, c) \
  __builtin_amdgcn_wmma_f32_16x16x32_f16(false, (a), false, (b), (short)0, (c), false, false)

// ---------------------------------------------------------------------------
// Kernel 1: fold weight(13,31,75) x basis(75,343) into per-tap f16 A-tiles,
// pre-swizzled into a contiguous-k 16x32 f16 A-operand lane layout (B uses the same k order) so the conv
// kernel does one contiguous 32B load per lane per tap.
//   lane<16 : M=lane,    element j -> K=j
//   lane>=16: M=lane-16, element j -> K=16+j
// ---------------------------------------------------------------------------
__global__ void prep_A_kernel(const float* __restrict__ W,
                              const float* __restrict__ Bas,
                              _Float16* __restrict__ Aw) {
  int idx = blockIdx.x * blockDim.x + threadIdx.x;
  if (idx >= NTAP * 512) return;
  int tap  = idx >> 9;
  int r    = idx & 511;
  int lane = r >> 4;
  int j    = r & 15;
  int vg   = j >> 1;
  int h    = j & 1;
  int m    = lane & 15;
  int kbase = (lane >> 4) * 16;
  int k    = kbase + 2 * vg + h;
  float v = 0.0f;
  if (m < CIN && k < CC) {
    const float* wr = W + ((size_t)m * CC + k) * NB;
    #pragma unroll 5
    for (int nb = 0; nb < NB; ++nb)
      v += wr[nb] * Bas[(size_t)nb * NTAP + tap];
  }
  Aw[idx] = (_Float16)v;
}

// ---------------------------------------------------------------------------
// Kernel 2: build zero-padded f16 xc in CHANNEL-FASTEST layout:
//   Xp[b][zp(54)][yp(54)][xp(54)][ch(32)], ch31 = 0 pad.
// Lane = channel -> each wave writes 64 contiguous bytes per voxel (coalesced).
// This makes the conv B operand a single contiguous 32B load per lane.
// ---------------------------------------------------------------------------
__global__ __launch_bounds__(256)
void prep_xc_kernel(const float* __restrict__ X, _Float16* __restrict__ Xp) {
  size_t idx = (size_t)blockIdx.x * blockDim.x + threadIdx.x;
  const size_t total = (size_t)BATCH * DP * DP * DP * KPAD;
  if (idx >= total) return;
  int ch = (int)(idx & (KPAD - 1));
  size_t vox = idx >> 5;
  int xp = (int)(vox % DP); size_t t1 = vox / DP;
  int yp = (int)(t1 % DP);  t1 /= DP;
  int zp = (int)(t1 % DP);
  int b  = (int)(t1 / DP);
  float v = 0.0f;
  if (ch < CC && zp >= 3 && zp < 51 && yp >= 3 && yp < 51 && xp >= 3 && xp < 51) {
    int z = zp - 3, y = yp - 3, xx = xp - 3;
    size_t sp = ((size_t)z * DD + y) * DD + xx;
    const float* xb = X + (size_t)b * CIN * PLANE;
    if (ch < CIN) {
      v = xb[(size_t)ch * PLANE + sp];
    } else {
      int c2 = ch - CIN;
      int ci = c2 / 9, rem = c2 % 9;
      int i3 = rem / 3, j3 = rem % 3;
      v = xb[(size_t)(2 + ci * 3 + i3) * PLANE + sp] *
          xb[(size_t)(2 + ci * 3 + j3) * PLANE + sp];
    }
  }
  Xp[idx] = (_Float16)v;
}

// ---------------------------------------------------------------------------
// Kernel 3: implicit-GEMM conv, LDS-free, with 2-row B reuse.
// One wave owns a 2-row x 48-voxel output slab (6 accumulators, N=96).
// Input row y0+u feeds output row y0 (tap dy=u) AND row y0+1 (tap dy=u-1),
// so each B tile load serves two WMMAs: per dz slice, 168 B loads + ~98 A
// loads (2nd touch = L0 hit) drive 294 v_wmma_f32_16x16x32_f16.
// All inner offsets are instruction immediates; xc + A-tiles are L2-resident.
// ---------------------------------------------------------------------------
__global__ __launch_bounds__(256)
void conv_kernel(const _Float16* __restrict__ Aw,
                 const _Float16* __restrict__ Xp,
                 float* __restrict__ Y) {
  const int tid  = threadIdx.x;
  const int wave = tid >> 5;
  const int lane = tid & 31;
  const int n    = lane & 15;
  const int hi   = lane >> 4;

  const int pairIdx = blockIdx.x * 8 + wave;     // 0 .. 4607
  const int b   = pairIdx / (DD * (DD / 2));
  const int rz  = pairIdx % (DD * (DD / 2));
  const int z   = rz / (DD / 2);
  const int y0  = (rz % (DD / 2)) * 2;           // even output row

  v8f acc00 = {}, acc01 = {}, acc02 = {};        // output row y0,   tiles 0..2
  v8f acc10 = {}, acc11 = {}, acc12 = {};        // output row y0+1, tiles 0..2

  // per-lane A base: lane*16 halves within a 512-half tap tile
  const _Float16* aBase = Aw + (size_t)lane * 16;
  // per-lane B base for (dz=0,u=0,dx=0, tile 0): voxel (z, y0, n), chunk hi
  const _Float16* xBase =
      Xp + ((((size_t)b * DP + z) * DP + y0) * DP + n) * KPAD + hi * 16;

  for (int dz = 0; dz < 7; ++dz) {
    #pragma unroll
    for (int u = 0; u < 8; ++u) {                // input row offset within slab
      const _Float16* xRow =
          xBase + ((size_t)dz * DP * DP + (size_t)u * DP) * KPAD;
      #pragma unroll
      for (int dx = 0; dx < 7; ++dx) {
        v16h b0 = *(const v16h*)(xRow + (size_t)(dx +  0) * KPAD);
        v16h b1 = *(const v16h*)(xRow + (size_t)(dx + 16) * KPAD);
        v16h b2 = *(const v16h*)(xRow + (size_t)(dx + 32) * KPAD);
        if (u < 7) {                             // row y0: tap dy = u
          v16h a0 = *(const v16h*)(aBase + (size_t)((dz * 7 + u) * 7 + dx) * 512);
          acc00 = WMMA_F16(a0, b0, acc00);
          acc01 = WMMA_F16(a0, b1, acc01);
          acc02 = WMMA_F16(a0, b2, acc02);
        }
        if (u > 0) {                             // row y0+1: tap dy = u-1
          v16h a1 = *(const v16h*)(aBase + (size_t)((dz * 7 + (u - 1)) * 7 + dx) * 512);
          acc10 = WMMA_F16(a1, b0, acc10);
          acc11 = WMMA_F16(a1, b1, acc11);
          acc12 = WMMA_F16(a1, b2, acc12);
        }
      }
    }
  }

  // ---- write pre-norm y (C/D layout: vgpr r -> M = r + 8*hi, N = lane&15) ----
  float* yBase = Y + (size_t)b * CIN * PLANE + (size_t)z * (DD * DD) + (size_t)y0 * DD;
  #pragma unroll
  for (int r = 0; r < 8; ++r) {
    int m = r + hi * 8;
    if (m < CIN) {
      float* yp = yBase + (size_t)m * PLANE + n;
      yp[0]       = acc00[r];
      yp[16]      = acc01[r];
      yp[32]      = acc02[r];
      yp[DD + 0]  = acc10[r];
      yp[DD + 16] = acc11[r];
      yp[DD + 32] = acc12[r];
    }
  }
}

// ---------------------------------------------------------------------------
// Kernel 4a/4b: global statistics.
//   stats[0..1] = sum / sumsq of channel 0     (scalar standardize)
//   stats[2..3] = sum / sumsq of channel 1
//   stats[4]    = sumsq of channels 2..4       (vector RMS group 0)
//   stats[5]    = sumsq of channels 5..7       (vector RMS group 1)
//   stats[6]    = sumsq of channels 8..12      (l=2 RMS group)
// ---------------------------------------------------------------------------
__global__ void zero_stats_kernel(float* s) {
  if (threadIdx.x < 8) s[threadIdx.x] = 0.0f;
}

__global__ __launch_bounds__(256)
void stats_kernel(const float* __restrict__ Y, float* __restrict__ S) {
  __shared__ float sh1[256];
  __shared__ float sh2[256];
  int blk   = blockIdx.x;                 // BATCH*13*108 blocks, 1024 elems each
  int chunk = blk % 108;
  int bc    = blk / 108;
  int c     = bc % CIN;
  const float* base = Y + (size_t)bc * PLANE + (size_t)chunk * 1024;
  float s = 0.0f, q = 0.0f;
  for (int i = threadIdx.x; i < 1024; i += 256) {
    float v = base[i];
    s += v; q += v * v;
  }
  sh1[threadIdx.x] = s; sh2[threadIdx.x] = q;
  __syncthreads();
  for (int off = 128; off > 0; off >>= 1) {
    if (threadIdx.x < off) {
      sh1[threadIdx.x] += sh1[threadIdx.x + off];
      sh2[threadIdx.x] += sh2[threadIdx.x + off];
    }
    __syncthreads();
  }
  if (threadIdx.x == 0) {
    if (c < 2) {
      atomicAdd(&S[c * 2 + 0], sh1[0]);
      atomicAdd(&S[c * 2 + 1], sh2[0]);
    } else if (c < 8) {
      atomicAdd(&S[4 + (c - 2) / 3], sh2[0]);
    } else {
      atomicAdd(&S[6], sh2[0]);
    }
  }
}

// ---------------------------------------------------------------------------
// Kernel 5: apply standardize / RMS-norm / bias / relu in place on d_out.
// ---------------------------------------------------------------------------
__global__ __launch_bounds__(256)
void apply_kernel(float* __restrict__ Y, const float* __restrict__ S,
                  const float* __restrict__ bias) {
  size_t idx = (size_t)blockIdx.x * blockDim.x + threadIdx.x;
  const size_t total = (size_t)BATCH * CIN * PLANE;
  if (idx >= total) return;
  int c = (int)((idx / PLANE) % CIN);
  float v = Y[idx];
  float out;
  if (c < 2) {
    const float inv_cnt = 1.0f / (BATCH * (float)PLANE);        // 1/442368
    float mean = S[c * 2 + 0] * inv_cnt;
    float var  = S[c * 2 + 1] * inv_cnt - mean * mean;
    out = (v - mean) * rsqrtf(var + EPS) + bias[c];
    out = fmaxf(out, 0.0f);
  } else if (c < 8) {
    const float inv_cnt = 1.0f / (BATCH * 3.0f * (float)PLANE); // 1/1327104
    out = v * rsqrtf(S[4 + (c - 2) / 3] * inv_cnt + EPS);
  } else {
    const float inv_cnt = 1.0f / (BATCH * 5.0f * (float)PLANE); // 1/2211840
    out = v * rsqrtf(S[6] * inv_cnt + EPS);
  }
  Y[idx] = out;
}

// ---------------------------------------------------------------------------
extern "C" void kernel_launch(void* const* d_in, const int* in_sizes, int n_in,
                              void* d_out, int out_size, void* d_ws, size_t ws_size,
                              hipStream_t stream) {
  const float* x      = (const float*)d_in[0];  // (4,13,48,48,48)
  const float* weight = (const float*)d_in[1];  // (13,31,75)
  const float* basis  = (const float*)d_in[2];  // (75,7,7,7)
  const float* bias   = (const float*)d_in[3];  // (2,)
  float* Y = (float*)d_out;                     // (4,13,48,48,48) f32

  // workspace layout
  char* ws = (char*)d_ws;
  const size_t off_A = 0;                                   // 343*512 f16
  const size_t sz_A  = (size_t)NTAP * 512 * sizeof(_Float16);
  const size_t off_X = off_A + sz_A;                        // 4*54^3*32 f16
  const size_t sz_X  = (size_t)BATCH * DP * DP * DP * KPAD * sizeof(_Float16);
  const size_t off_S = off_X + sz_X;                        // 8 f32 stats
  _Float16* Aw    = (_Float16*)(ws + off_A);
  _Float16* Xp    = (_Float16*)(ws + off_X);
  float*    stats = (float*)(ws + off_S);

  // 1) fold weight x basis into per-tap WMMA A-tiles
  {
    int total = NTAP * 512;
    prep_A_kernel<<<(total + 255) / 256, 256, 0, stream>>>(weight, basis, Aw);
  }
  // 2) build padded f16 xc, channel-fastest (L2-resident: ~40 MB of 192 MB)
  {
    size_t total = (size_t)BATCH * DP * DP * DP * KPAD;
    prep_xc_kernel<<<(unsigned)((total + 255) / 256), 256, 0, stream>>>(x, Xp);
  }
  // 3) zero stats
  zero_stats_kernel<<<1, 32, 0, stream>>>(stats);
  // 4) LDS-free WMMA implicit-GEMM conv (2-row slab per wave) -> d_out
  conv_kernel<<<(BATCH * DD * (DD / 2)) / 8, 256, 0, stream>>>(Aw, Xp, Y);
  // 5) global statistics
  stats_kernel<<<BATCH * CIN * 108, 256, 0, stream>>>(Y, stats);
  // 6) normalize + bias + relu in place
  {
    size_t total = (size_t)BATCH * CIN * PLANE;
    apply_kernel<<<(unsigned)((total + 255) / 256), 256, 0, stream>>>(Y, stats, bias);
  }
}
